// SlowMoE_25615184953559
// MI455X (gfx1250) — matrix operand order repaired and checked by reading, not compiled, not run on hardware
//
#include <hip/hip_runtime.h>
#include <hip/hip_bf16.h>

// ---------------------------------------------------------------------------
// MoE forward (router + grouped routed SwiGLU experts + shared SwiGLU expert)
// for MI455X / gfx1250: bf16 WMMA (V_WMMA_F32_16X16X32_BF16), wave32,
// double-buffered LDS staging via GLOBAL_LOAD_ASYNC_TO_LDS_B128 (ASYNCcnt).
// ---------------------------------------------------------------------------

typedef __bf16 bf16_t;
typedef bf16_t v4bf  __attribute__((ext_vector_type(4)));
typedef bf16_t v8bf  __attribute__((ext_vector_type(8)));
typedef bf16_t v16bf __attribute__((ext_vector_type(16)));
typedef float  v4f   __attribute__((ext_vector_type(4)));
typedef float  v8f   __attribute__((ext_vector_type(8)));

static constexpr int Bsz = 2, Ssz = 2048, Hdim = 2048, Edim = 8, Idim = 1408;
static constexpr int Tdim = Bsz * Ssz;        // 4096 tokens
static constexpr int SI   = Idim * 2;         // 2816 shared intermediate
static constexpr int TK   = Tdim * 2;         // 8192 routed (token, k) rows

static constexpr int BM = 64, BN = 64, BK = 32, NTHR = 128;
static constexpr int LDA_S = 40;  // padded bf16 row stride for 64x32 A tile
static constexpr int LDB_S = 72;  // padded bf16 row stride for 32x64 B tile

__device__ __forceinline__ v16bf cat8(v8bf lo, v8bf hi) {
  return __builtin_shufflevector(lo, hi, 0,1,2,3,4,5,6,7,8,9,10,11,12,13,14,15);
}

// Async global->LDS copy of 16 bytes per lane (tracked by ASYNCcnt).
// %0 = LDS byte address (low 32 bits of flat shared pointer),
// %1 = 64-bit global address.
__device__ __forceinline__ void async_cp16(unsigned ldsAddr, const bf16_t* g) {
  asm volatile("global_load_async_to_lds_b128 %0, %1, off"
               :: "v"(ldsAddr), "v"(g) : "memory");
}

__device__ __forceinline__ unsigned lds_addr_of(const void* p) {
  return (unsigned)(unsigned long long)p;   // LDS aperture: addr[31:0] = LDS offset
}

// ---------------------------------------------------------------------------
// Grouped GEMM:  D[row, n] = A[src(row), :] @ B[:, n]   (bf16 x bf16 -> f32 acc)
// SWIGLU=true: two B matrices (w1,w3), epilogue silu(h)*g -> bf16 out.
// groupOff != null: blockIdx.z picks expert group [off[z], off[z+1]);
// gatherIdx != null: A row indices are gathered token ids.
// All A/B operands are bf16 in global memory (weights pre-converted).
// ---------------------------------------------------------------------------
template <bool SWIGLU>
__global__ __launch_bounds__(NTHR)
void gemm_bf16(const bf16_t* __restrict__ A, int lda,
               const int* __restrict__ gatherIdx,
               const int* __restrict__ groupOff, int totalRows,
               const bf16_t* __restrict__ W1g, const bf16_t* __restrict__ W3g,
               long wstride, int Kdim, int ldb,
               bf16_t* __restrict__ outBf, float* __restrict__ outF, int ldo)
{
  __shared__ __align__(16) bf16_t sA [2][BM * LDA_S];
  __shared__ __align__(16) bf16_t sB0[2][BK * LDB_S];
  __shared__ __align__(16) bf16_t sB1[SWIGLU ? 2 : 1][BK * LDB_S];

  const int e = blockIdx.z;
  int rowBeg, rowEnd;
  if (groupOff) { rowBeg = groupOff[e]; rowEnd = groupOff[e + 1]; }
  else          { rowBeg = 0;           rowEnd = totalRows; }
  const int base = rowBeg + blockIdx.y * BM;
  if (base >= rowEnd) return;                 // grouped-GEMM early exit
  const int n0 = blockIdx.x * BN;

  const bf16_t* __restrict__ W1 = W1g + (long)e * wstride;
  const bf16_t* __restrict__ W3 = SWIGLU ? (W3g + (long)e * wstride) : nullptr;

  const int tid = threadIdx.x;

  // ---- A staging: 256 16B chunks; thread owns (r0,q) and (r0+32,q).
  const int r0 = tid >> 2, q = tid & 3;
  long aoff0, aoff1;
  {
    int g0 = base + r0;      if (g0 > rowEnd - 1) g0 = rowEnd - 1;
    int g1 = base + r0 + 32; if (g1 > rowEnd - 1) g1 = rowEnd - 1;
    int a0 = gatherIdx ? gatherIdx[g0] : g0;
    int a1 = gatherIdx ? gatherIdx[g1] : g1;
    aoff0 = (long)a0 * lda + q * 8;
    aoff1 = (long)a1 * lda + q * 8;
  }
  const unsigned aDstOff0 = (unsigned)((r0      * LDA_S + q * 8) * 2);
  const unsigned aDstOff1 = (unsigned)(((r0+32) * LDA_S + q * 8) * 2);

  // ---- B staging: 256 16B chunks per matrix; thread owns rows bk, bk+16, col g8*8.
  const int bk = tid >> 3, g8 = tid & 7;
  const unsigned bDstOff0 = (unsigned)((bk        * LDB_S + g8 * 8) * 2);
  const unsigned bDstOff1 = (unsigned)(((bk + 16) * LDB_S + g8 * 8) * 2);

  unsigned sAb[2]  = { lds_addr_of(&sA[0][0]),  lds_addr_of(&sA[1][0])  };
  unsigned sB0b[2] = { lds_addr_of(&sB0[0][0]), lds_addr_of(&sB0[1][0]) };
  unsigned sB1b[2] = { lds_addr_of(&sB1[0][0]),
                       lds_addr_of(&sB1[SWIGLU ? 1 : 0][0]) };

  auto stage = [&](int buf, int k0) {
    async_cp16(sAb[buf] + aDstOff0, A + aoff0 + k0);
    async_cp16(sAb[buf] + aDstOff1, A + aoff1 + k0);
    const bf16_t* b0 = W1 + (long)k0 * ldb + n0 + g8 * 8;
    async_cp16(sB0b[buf] + bDstOff0, b0 + (long)bk * ldb);
    async_cp16(sB0b[buf] + bDstOff1, b0 + (long)(bk + 16) * ldb);
    if constexpr (SWIGLU) {
      const bf16_t* b1 = W3 + (long)k0 * ldb + n0 + g8 * 8;
      async_cp16(sB1b[buf] + bDstOff0, b1 + (long)bk * ldb);
      async_cp16(sB1b[buf] + bDstOff1, b1 + (long)(bk + 16) * ldb);
    }
  };

  // ---- compute geometry: 4 waves in 2x2 grid, each wave a 32x32 tile.
  const int lane = tid & 31, wave = tid >> 5;
  const int wm = (wave & 1) * 32, wn = (wave >> 1) * 32;
  const int lh = lane >> 4, lm = lane & 15;

  v8f hacc[2][2] = {};
  v8f gacc[2][2] = {};

  stage(0, 0);                         // prologue: fill buffer 0
  for (int k0 = 0; k0 < Kdim; k0 += BK) {
    const int cur = (k0 / BK) & 1;
    if (k0 + BK < Kdim) {
      stage(cur ^ 1, k0 + BK);         // issue next buffer's copies
      // wait until only the just-issued copies remain -> cur buffer landed
      asm volatile("s_wait_asynccnt %0" :: "i"(SWIGLU ? 6 : 4) : "memory");
    } else {
      asm volatile("s_wait_asynccnt 0" ::: "memory");
    }
    __syncthreads();                   // all waves' cur copies visible

    // fragments: A lane=M, K packed along VGPRs (lane-half h holds K 8h..8h+7, 16+8h..23+8h);
    //            B lane=N, K packed along VGPRs (same K order as A).
    v16bf af[2], b0f[2], b1f[2];
#pragma unroll
    for (int mi = 0; mi < 2; ++mi) {
      const bf16_t* p = &sA[cur][(wm + mi * 16 + lm) * LDA_S + lh * 8];
      af[mi] = cat8(*(const v8bf*)p, *(const v8bf*)(p + 16));
    }
#pragma unroll
    for (int ni = 0; ni < 2; ++ni) {
      const bf16_t* p = &sB0[cur][(lh * 8) * LDB_S + wn + ni * 16 + lm];
#pragma unroll
      for (int s = 0; s < 16; ++s) b0f[ni][s] = p[(s < 8 ? s : s + 8) * LDB_S];
      if constexpr (SWIGLU) {
        const bf16_t* p1 = &sB1[cur][(lh * 8) * LDB_S + wn + ni * 16 + lm];
#pragma unroll
        for (int s = 0; s < 16; ++s) b1f[ni][s] = p1[(s < 8 ? s : s + 8) * LDB_S];
      }
    }
#pragma unroll
    for (int mi = 0; mi < 2; ++mi)
#pragma unroll
      for (int ni = 0; ni < 2; ++ni) {
        hacc[mi][ni] = __builtin_amdgcn_wmma_f32_16x16x32_bf16(
            false, af[mi], false, b0f[ni], (short)0, hacc[mi][ni], false, false);
        if constexpr (SWIGLU)
          gacc[mi][ni] = __builtin_amdgcn_wmma_f32_16x16x32_bf16(
              false, af[mi], false, b1f[ni], (short)0, gacc[mi][ni], false, false);
      }
    __syncthreads();                   // protect cur buffer from next issue
  }

  // epilogue; C layout: VGPR r -> M = r + 8*(lane/16), N = lane%16
#pragma unroll
  for (int mi = 0; mi < 2; ++mi)
#pragma unroll
    for (int ni = 0; ni < 2; ++ni) {
      const int rb = base + wm + mi * 16 + lh * 8;
      const int cb = n0 + wn + ni * 16 + lm;
#pragma unroll
      for (int r = 0; r < 8; ++r) {
        const int row = rb + r;
        if (row >= rowEnd) continue;
        if constexpr (SWIGLU) {
          float h = hacc[mi][ni][r], g = gacc[mi][ni][r];
          float s = h / (1.0f + __expf(-h));          // silu
          outBf[(long)row * ldo + cb] = (bf16_t)(s * g);
        } else {
          outF[(long)row * ldo + cb] = hacc[mi][ni][r];
        }
      }
    }
}

// ---------------------------------------------------------------------------
// Small helper kernels
// ---------------------------------------------------------------------------
__global__ void zero_counts(int* cnt, int* cur) {
  int i = threadIdx.x;
  if (i < Edim) { cnt[i] = 0; cur[i] = 0; }
}

__global__ void f32_to_bf16(const float* __restrict__ x, bf16_t* __restrict__ xb, long n) {
  long i = ((long)blockIdx.x * blockDim.x + threadIdx.x) * 4;
  if (i >= n) return;
  v4f v = *(const v4f*)(x + i);
  v4bf p; p[0] = (bf16_t)v[0]; p[1] = (bf16_t)v[1];
          p[2] = (bf16_t)v[2]; p[3] = (bf16_t)v[3];
  *(v4bf*)(xb + i) = p;
}

__global__ __launch_bounds__(256)
void router_kernel(const float* __restrict__ x, const float* __restrict__ gw,
                   int* __restrict__ topkIdx, float* __restrict__ topkW,
                   int* __restrict__ cnt)
{
  __shared__ float red[Edim * 256];
  const int t = blockIdx.x, tid = threadIdx.x;
  float p[Edim];
#pragma unroll
  for (int e = 0; e < Edim; ++e) p[e] = 0.f;
  for (int h = tid; h < Hdim; h += 256) {
    float xv = x[(long)t * Hdim + h];
#pragma unroll
    for (int e = 0; e < Edim; ++e) p[e] += xv * gw[e * Hdim + h];
  }
#pragma unroll
  for (int e = 0; e < Edim; ++e) red[e * 256 + tid] = p[e];
  __syncthreads();
  for (int s = 128; s > 0; s >>= 1) {
    if (tid < s) {
#pragma unroll
      for (int e = 0; e < Edim; ++e) red[e * 256 + tid] += red[e * 256 + tid + s];
    }
    __syncthreads();
  }
  if (tid == 0) {
    float l[Edim], mx = -1e30f;
    for (int e = 0; e < Edim; ++e) { l[e] = red[e * 256]; mx = fmaxf(mx, l[e]); }
    float sum = 0.f;
    for (int e = 0; e < Edim; ++e) { l[e] = __expf(l[e] - mx); sum += l[e]; }
    float inv = 1.f / sum;
    int i0 = 0; float s0 = -1.f;
    for (int e = 0; e < Edim; ++e) { float sc = l[e] * inv; if (sc > s0) { s0 = sc; i0 = e; } }
    int i1 = 0; float s1 = -1.f;
    for (int e = 0; e < Edim; ++e) { if (e == i0) continue;
      float sc = l[e] * inv; if (sc > s1) { s1 = sc; i1 = e; } }
    float nrm = 1.f / (s0 + s1);
    topkIdx[t * 2] = i0; topkIdx[t * 2 + 1] = i1;
    topkW[t * 2] = s0 * nrm; topkW[t * 2 + 1] = s1 * nrm;
    atomicAdd(&cnt[i0], 1); atomicAdd(&cnt[i1], 1);
  }
}

__global__ void scan_kernel(const int* cnt, int* off, int* cur) {
  if (threadIdx.x == 0 && blockIdx.x == 0) {
    int a = 0;
    for (int e = 0; e < Edim; ++e) { off[e] = a; a += cnt[e]; cur[e] = 0; }
    off[Edim] = a;
  }
}

__global__ void place_kernel(const int* __restrict__ topkIdx, const int* __restrict__ off,
                             int* __restrict__ cur, int* __restrict__ gidx,
                             int* __restrict__ slotOf)
{
  int t = blockIdx.x * blockDim.x + threadIdx.x;
  if (t >= Tdim) return;
#pragma unroll
  for (int k = 0; k < 2; ++k) {
    int e = topkIdx[t * 2 + k];
    int s = off[e] + atomicAdd(&cur[e], 1);
    gidx[s] = t;
    slotOf[t * 2 + k] = s;
  }
}

__global__ __launch_bounds__(256)
void combine_kernel(const float* __restrict__ yslot, const float* __restrict__ sout,
                    const int* __restrict__ slotOf, const float* __restrict__ topkW,
                    float* __restrict__ out)
{
  const int t = blockIdx.y;
  const int h = blockIdx.x * 256 + threadIdx.x;
  const int s0 = slotOf[t * 2], s1 = slotOf[t * 2 + 1];
  const float w0 = topkW[t * 2], w1 = topkW[t * 2 + 1];
  out[(long)t * Hdim + h] = w0 * yslot[(long)s0 * Hdim + h]
                          + w1 * yslot[(long)s1 * Hdim + h]
                          + sout[(long)t * Hdim + h];
}

// ---------------------------------------------------------------------------
extern "C" void kernel_launch(void* const* d_in, const int* in_sizes, int n_in,
                              void* d_out, int out_size, void* d_ws, size_t ws_size,
                              hipStream_t stream)
{
  const float* x   = (const float*)d_in[0];
  const float* gw  = (const float*)d_in[1];
  const float* w1  = (const float*)d_in[2];
  const float* w3  = (const float*)d_in[3];
  const float* w2  = (const float*)d_in[4];
  const float* sw1 = (const float*)d_in[5];
  const float* sw3 = (const float*)d_in[6];
  const float* sw2 = (const float*)d_in[7];
  float* out = (float*)d_out;

  // workspace carve (~337 MB total: activations + bf16 weight copies)
  char* w = (char*)d_ws;
  auto take = [&](size_t bytes) -> void* {
    void* p = (void*)w;
    w += (bytes + 255) & ~size_t(255);
    return p;
  };
  bf16_t* xb     = (bf16_t*)take((size_t)Tdim * Hdim * 2);
  bf16_t* act    = (bf16_t*)take((size_t)TK * Idim * 2);
  bf16_t* acts   = (bf16_t*)take((size_t)Tdim * SI * 2);
  float*  yslot  = (float*)take((size_t)TK * Hdim * 4);
  float*  sout   = (float*)take((size_t)Tdim * Hdim * 4);
  bf16_t* wb1    = (bf16_t*)take((size_t)Edim * Hdim * Idim * 2);
  bf16_t* wb3    = (bf16_t*)take((size_t)Edim * Hdim * Idim * 2);
  bf16_t* wb2    = (bf16_t*)take((size_t)Edim * Idim * Hdim * 2);
  bf16_t* swb1   = (bf16_t*)take((size_t)Hdim * SI * 2);
  bf16_t* swb3   = (bf16_t*)take((size_t)Hdim * SI * 2);
  bf16_t* swb2   = (bf16_t*)take((size_t)SI * Hdim * 2);
  int*    tkI    = (int*)take((size_t)Tdim * 2 * 4);
  float*  tkW    = (float*)take((size_t)Tdim * 2 * 4);
  int*    gidx   = (int*)take((size_t)TK * 4);
  int*    slotOf = (int*)take((size_t)TK * 4);
  int*    cnt    = (int*)take(64);
  int*    cur    = (int*)take(64);
  int*    off    = (int*)take(64);

  auto conv = [&](const float* src, bf16_t* dst, long n) {
    f32_to_bf16<<<(unsigned)(n / 4 / 256), 256, 0, stream>>>(src, dst, n);
  };

  zero_counts<<<1, 32, 0, stream>>>(cnt, cur);
  conv(x,   xb,   (long)Tdim * Hdim);
  conv(w1,  wb1,  (long)Edim * Hdim * Idim);
  conv(w3,  wb3,  (long)Edim * Hdim * Idim);
  conv(w2,  wb2,  (long)Edim * Idim * Hdim);
  conv(sw1, swb1, (long)Hdim * SI);
  conv(sw3, swb3, (long)Hdim * SI);
  conv(sw2, swb2, (long)SI * Hdim);
  router_kernel<<<Tdim, 256, 0, stream>>>(x, gw, tkI, tkW, cnt);
  scan_kernel<<<1, 1, 0, stream>>>(cnt, off, cur);
  place_kernel<<<Tdim / 256, 256, 0, stream>>>(tkI, off, cur, gidx, slotOf);

  // routed stage 1: act[slot, I] = silu(Xg @ w1[e]) * (Xg @ w3[e])
  gemm_bf16<true><<<dim3(Idim / BN, Tdim / BM, Edim), NTHR, 0, stream>>>(
      xb, Hdim, gidx, off, 0,
      wb1, wb3, (long)Hdim * Idim, Hdim, Idim,
      act, nullptr, Idim);
  // shared stage 1: acts[t, SI] = silu(X @ sw1) * (X @ sw3)
  gemm_bf16<true><<<dim3(SI / BN, Tdim / BM, 1), NTHR, 0, stream>>>(
      xb, Hdim, nullptr, nullptr, Tdim,
      swb1, swb3, 0, Hdim, SI,
      acts, nullptr, SI);
  // routed stage 2: yslot[slot, H] = act @ w2[e]
  gemm_bf16<false><<<dim3(Hdim / BN, Tdim / BM, Edim), NTHR, 0, stream>>>(
      act, Idim, nullptr, off, 0,
      wb2, nullptr, (long)Idim * Hdim, Idim, Hdim,
      nullptr, yslot, Hdim);
  // shared stage 2: sout[t, H] = acts @ sw2
  gemm_bf16<false><<<dim3(Hdim / BN, Tdim / BM, 1), NTHR, 0, stream>>>(
      acts, SI, nullptr, nullptr, Tdim,
      swb2, nullptr, 0, SI, Hdim,
      nullptr, sout, Hdim);

  combine_kernel<<<dim3(Hdim / 256, Tdim), 256, 0, stream>>>(yslot, sout, slotOf, tkW, out);
}
